// Conv1D_MEO_88055419502754
// MI455X (gfx1250) — matrix operand order repaired and checked by reading, not compiled, not run on hardware
//
#include <hip/hip_runtime.h>
#include <hip/hip_bf16.h>

// ---------------------------------------------------------------------------
// Problem constants (from reference): E=8, IN=OUT=1024, B=4, L=2048, T=256
// BN = B*L/T = 32 token groups, RANK = 8.
// ---------------------------------------------------------------------------
#define E_EXP   8
#define DIN     1024
#define DOUT    1024
#define BN      32
#define TGRP    256
#define RANK    8
#define YSZ     (BN * TGRP * DOUT)      // 8,388,608 floats; loss at d_out[YSZ]

typedef float v8f  __attribute__((ext_vector_type(8)));
typedef __bf16 v16bf __attribute__((ext_vector_type(16)));
typedef int v4i __attribute__((ext_vector_type(4)));

#if __has_builtin(__builtin_amdgcn_global_load_async_to_lds_b128) && \
    __has_builtin(__builtin_amdgcn_s_wait_asynccnt)
#define USE_ASYNC 1
#else
#define USE_ASYNC 0
#endif

__device__ __forceinline__ unsigned short f2bf(float f) {
  unsigned u = __builtin_bit_cast(unsigned, f);
  unsigned r = u + 0x7FFFu + ((u >> 16) & 1u);   // round-to-nearest-even
  return (unsigned short)(r >> 16);
}
__device__ __forceinline__ float sgnf(float x) {
  return (x > 0.f) ? 1.f : ((x < 0.f) ? -1.f : 0.f);
}

// Memory -> LDS tile mover: async on CDNA5 (ASYNCcnt), sync fallback otherwise.
__device__ __forceinline__ void tile_b128(const unsigned short* g, unsigned short* l) {
#if USE_ASYNC
  __builtin_amdgcn_global_load_async_to_lds_b128(
      (__attribute__((address_space(1))) v4i*)g,
      (__attribute__((address_space(3))) v4i*)l, 0, 0);
#else
  *(uint4*)l = *(const uint4*)g;
#endif
}
__device__ __forceinline__ void tile_wait() {
#if USE_ASYNC
  __builtin_amdgcn_s_wait_asynccnt(0);
#endif
}

// ---------------------------------------------------------------------------
// 1. Group means of x over T tokens:  colmean[bn][c] = mean_t x[bn*T+t][c]
// ---------------------------------------------------------------------------
__global__ __launch_bounds__(256) void colmean_kernel(const float* __restrict__ x,
                                                      float* __restrict__ colmean) {
  int bn = blockIdx.x, tid = threadIdx.x;
  float acc[4] = {0.f, 0.f, 0.f, 0.f};
  for (int t = 0; t < TGRP; ++t) {
    const float* row = x + ((size_t)(bn * TGRP + t)) * DIN;
#pragma unroll
    for (int k = 0; k < 4; ++k) acc[k] += row[tid + k * 256];
  }
#pragma unroll
  for (int k = 0; k < 4; ++k)
    colmean[bn * DIN + tid + k * 256] = acc[k] * (1.0f / TGRP);
}

// ---------------------------------------------------------------------------
// 2. Gating: logits, softmax, importance/load CV loss, roll (single block)
// ---------------------------------------------------------------------------
__global__ __launch_bounds__(256) void gate_kernel(const float* __restrict__ colmean,
                                                   const float* __restrict__ w_gate,
                                                   float* __restrict__ gates_rolled,
                                                   float* __restrict__ loss_out) {
  __shared__ float lg[BN * E_EXP];
  __shared__ float gg[BN * E_EXP];
  __shared__ float imp[E_EXP];
  __shared__ float ldc[E_EXP];
  int tid = threadIdx.x;
  int bn = tid >> 3, e = tid & 7;
  float s = 0.f;
  for (int d = 0; d < DIN; ++d) s += colmean[bn * DIN + d] * w_gate[d * E_EXP + e];
  lg[tid] = s;
  __syncthreads();
  float m = -1e30f;
#pragma unroll
  for (int k = 0; k < E_EXP; ++k) m = fmaxf(m, lg[bn * E_EXP + k]);
  float den = 0.f;
#pragma unroll
  for (int k = 0; k < E_EXP; ++k) den += __expf(lg[bn * E_EXP + k] - m);
  float g = __expf(lg[tid] - m) / den;
  gg[tid] = g;
  __syncthreads();
  if (tid < E_EXP) {
    float si = 0.f, sl = 0.f;
    for (int b = 0; b < BN; ++b) {
      float v = gg[b * E_EXP + tid];
      si += v;
      sl += (v > 0.f) ? 1.f : 0.f;
    }
    imp[tid] = si; ldc[tid] = sl;
  }
  __syncthreads();
  if (tid == 0) {
    float mi = 0.f, ml = 0.f;
    for (int k = 0; k < E_EXP; ++k) { mi += imp[k]; ml += ldc[k]; }
    mi /= E_EXP; ml /= E_EXP;
    float vi = 0.f, vl = 0.f;
    for (int k = 0; k < E_EXP; ++k) {
      vi += (imp[k] - mi) * (imp[k] - mi);
      vl += (ldc[k] - ml) * (ldc[k] - ml);
    }
    vi /= (E_EXP - 1); vl /= (E_EXP - 1);
    float cvi = vi / (mi * mi + 1e-10f);
    float cvl = vl / (ml * ml + 1e-10f);
    *loss_out = (cvi + cvl) * 1e-3f;
  }
  // roll by one group within flattened BN, keep first group of each batch
  int src = ((bn & 7) == 0) ? tid : tid - E_EXP;
  gates_rolled[tid] = gg[src];
}

// ---------------------------------------------------------------------------
// 3. Deterministic init of subspace Q[e][o][r]
// ---------------------------------------------------------------------------
__global__ __launch_bounds__(256) void initq_kernel(float* __restrict__ Q) {
  int idx = blockIdx.x * 256 + threadIdx.x;     // 8*1024*8 = 65536
  int r = idx & 7, o = (idx >> 3) & 1023;
  float v = ((o & 127) == (r * 16)) ? 1.0f : 0.0f;
  v += 1e-4f * (float)((o * 131 + r * 29) % 251);
  Q[idx] = v;
}

// ---------------------------------------------------------------------------
// 4a. t1[e][i][r] = sum_o (weight[e,o,i]-res[o,i]) * Q[e,o,r]   (= D^T Q)
//     (also reused at the end as rtw0 transposed: rtw0[e,r,i] = t1[e,i,r])
// ---------------------------------------------------------------------------
__global__ __launch_bounds__(256) void t1_kernel(const float* __restrict__ weight,
                                                 const float* __restrict__ resw,
                                                 const float* __restrict__ Q,
                                                 float* __restrict__ t1) {
  __shared__ float Qs[DOUT * RANK];             // 32 KB
  int e = blockIdx.y;
  int i = blockIdx.x * 256 + threadIdx.x;
  for (int m = 0; m < 32; ++m)
    Qs[threadIdx.x + 256 * m] = Q[e * DOUT * RANK + threadIdx.x + 256 * m];
  __syncthreads();
  float acc[RANK];
#pragma unroll
  for (int r = 0; r < RANK; ++r) acc[r] = 0.f;
  const float* wcol = weight + (size_t)e * DOUT * DIN + i;
  for (int o = 0; o < DOUT; ++o) {
    float d = wcol[(size_t)o * DIN] - resw[o * DIN + i];
#pragma unroll
    for (int r = 0; r < RANK; ++r) acc[r] += d * Qs[o * RANK + r];
  }
#pragma unroll
  for (int r = 0; r < RANK; ++r) t1[e * DOUT * RANK + i * RANK + r] = acc[r];
}

// 4b. Z[e][o][r] = sum_i (weight[e,o,i]-res[o,i]) * t1[e,i,r]   (= D (D^T Q))
__global__ __launch_bounds__(256) void z_kernel(const float* __restrict__ weight,
                                                const float* __restrict__ resw,
                                                const float* __restrict__ t1,
                                                float* __restrict__ Z) {
  __shared__ float t1s[DIN * RANK];             // 32 KB
  int e = blockIdx.y;
  int o = blockIdx.x * 256 + threadIdx.x;
  for (int m = 0; m < 32; ++m)
    t1s[threadIdx.x + 256 * m] = t1[e * DIN * RANK + threadIdx.x + 256 * m];
  __syncthreads();
  float acc[RANK];
#pragma unroll
  for (int r = 0; r < RANK; ++r) acc[r] = 0.f;
  const float* wrow = weight + (size_t)e * DOUT * DIN + (size_t)o * DIN;
  const float* rrow = resw + (size_t)o * DIN;
  for (int i = 0; i < DIN; ++i) {
    float d = wrow[i] - rrow[i];
#pragma unroll
    for (int r = 0; r < RANK; ++r) acc[r] += d * t1s[i * RANK + r];
  }
#pragma unroll
  for (int r = 0; r < RANK; ++r) Z[e * DOUT * RANK + o * RANK + r] = acc[r];
}

// 4c. Modified Gram-Schmidt: Q[e] = orth(Z[e]), one block per expert
__device__ __forceinline__ float block_reduce(float v, float* red) {
  int t = threadIdx.x;
  red[t] = v;
  __syncthreads();
  for (int s = 128; s > 0; s >>= 1) {
    if (t < s) red[t] += red[t + s];
    __syncthreads();
  }
  float r = red[0];
  __syncthreads();
  return r;
}

__global__ __launch_bounds__(256) void mgs_kernel(float* __restrict__ Z,
                                                  float* __restrict__ Q) {
  __shared__ float red[256];
  __shared__ float col[DOUT];
  int e = blockIdx.x, tid = threadIdx.x;
  float* Ze = Z + e * DOUT * RANK;
  float* Qe = Q + e * DOUT * RANK;
  for (int j = 0; j < RANK; ++j) {
    float s = 0.f;
    for (int m = 0; m < 4; ++m) {
      int o = tid + 256 * m;
      float v = Ze[o * RANK + j];
      s += v * v;
    }
    float nrm = block_reduce(s, red);
    float inv = rsqrtf(nrm + 1e-20f);
    for (int m = 0; m < 4; ++m) {
      int o = tid + 256 * m;
      float qv = Ze[o * RANK + j] * inv;
      col[o] = qv;
      Qe[o * RANK + j] = qv;
    }
    __syncthreads();
    for (int k = j + 1; k < RANK; ++k) {
      float d = 0.f;
      for (int m = 0; m < 4; ++m) {
        int o = tid + 256 * m;
        d += col[o] * Ze[o * RANK + k];
      }
      float dot = block_reduce(d, red);
      for (int m = 0; m < 4; ++m) {
        int o = tid + 256 * m;
        Ze[o * RANK + k] -= dot * col[o];
      }
      __syncthreads();
    }
  }
}

// ---------------------------------------------------------------------------
// 5. Kronecker curve transforms on rank core [E, 2,4,32,32]
//    stage 1 reads t1 ([e][i][r] layout); stages 2-4 read [e][r][i] layout.
// ---------------------------------------------------------------------------
__global__ __launch_bounds__(256) void curve_kernel(const float* __restrict__ in,
                                                    const float* __restrict__ cmat,
                                                    float* __restrict__ out,
                                                    int stage) {
  int idx = blockIdx.x * 256 + threadIdx.x;     // 0..8191
  int e = blockIdx.y;
  int m = idx & 31, l = (idx >> 5) & 31, k = (idx >> 10) & 3, j = idx >> 12;
  float s = 0.f;
  if (stage == 1) {        // out[i2,k,l,m] = sum_j c1o[i2,j]*in[j,k,l,m]
#pragma unroll
    for (int jj = 0; jj < 2; ++jj)
      s += cmat[e * 4 + j * 2 + jj] * in[e * 8192 + (l * 32 + m) * 8 + jj * 4 + k];
  } else if (stage == 2) { // out[j,i4,l,m] = sum_k c2o[i4,k]*in[j,k,l,m]
#pragma unroll
    for (int kk = 0; kk < 4; ++kk)
      s += cmat[e * 16 + k * 4 + kk] * in[e * 8192 + ((j * 4 + kk) * 32 + l) * 32 + m];
  } else if (stage == 3) { // out[j,k,i32,m] = sum_l c1i[i32,l]*in[j,k,l,m]
    for (int ll = 0; ll < 32; ++ll)
      s += cmat[e * 1024 + l * 32 + ll] * in[e * 8192 + ((j * 4 + k) * 32 + ll) * 32 + m];
  } else {                 // out[j,k,l,i32] = sum_m c2i[i32,m]*in[j,k,l,m]
    for (int mm = 0; mm < 32; ++mm)
      s += cmat[e * 1024 + m * 32 + mm] * in[e * 8192 + ((j * 4 + k) * 32 + l) * 32 + mm];
  }
  out[e * 8192 + idx] = s;
}

// ---------------------------------------------------------------------------
// 6. Bias curve path: rtb2[e][i*32+k] = sum_j (sum_i' c1b[k,i']*db[i',j]) * c2b[i,j]
// ---------------------------------------------------------------------------
__global__ __launch_bounds__(256) void bias1_kernel(const float* __restrict__ bias,
                                                    const float* __restrict__ res_bias,
                                                    const float* __restrict__ c1b,
                                                    const float* __restrict__ c2b,
                                                    float* __restrict__ rtb2) {
  __shared__ float db[1024];
  __shared__ float o1[1024];
  int e = blockIdx.x, tid = threadIdx.x;
  for (int m = 0; m < 4; ++m) {
    int o = tid + 256 * m;
    db[o] = bias[e * DOUT + o] - res_bias[o];
  }
  __syncthreads();
  for (int m = 0; m < 4; ++m) {
    int idx = tid + 256 * m;
    int kk = idx >> 5, jj = idx & 31;
    float s = 0.f;
    for (int ii = 0; ii < 32; ++ii)
      s += c1b[e * 1024 + kk * 32 + ii] * db[ii * 32 + jj];
    o1[idx] = s;
  }
  __syncthreads();
  for (int m = 0; m < 4; ++m) {
    int idx = tid + 256 * m;
    int ii = idx >> 5, kk = idx & 31;
    float s = 0.f;
    for (int jj = 0; jj < 32; ++jj)
      s += o1[kk * 32 + jj] * c2b[e * 1024 + ii * 32 + jj];
    rtb2[e * DOUT + idx] = s;                   // layout i*32+k
  }
}

// 7. expert_bias[b][o] = res_bias[o] + sum_e g[b,e]*rtb2[e,o]*mask_b[e,o]
__global__ __launch_bounds__(256) void ebias_kernel(const float* __restrict__ bias,
                                                    const float* __restrict__ res_bias,
                                                    const float* __restrict__ rtb2,
                                                    const float* __restrict__ gates,
                                                    float* __restrict__ ebias) {
  __shared__ float gls[BN * E_EXP];
  int tid = threadIdx.x;
  int o = blockIdx.x * 256 + tid;
  gls[tid] = gates[tid];
  __syncthreads();
  float rb = res_bias[o];
  float diff[E_EXP], val[E_EXP], S = 0.f;
#pragma unroll
  for (int e = 0; e < E_EXP; ++e) {
    diff[e] = bias[e * DOUT + o] - rb;
    S += diff[e];
  }
  float sS = sgnf(S);
#pragma unroll
  for (int e = 0; e < E_EXP; ++e) {
    float sd = sgnf(diff[e]);
    float mask = (sd == sS) ? sd : 0.f;
    val[e] = rtb2[e * DOUT + o] * mask;
  }
  for (int b = 0; b < BN; ++b) {
    float s = rb;
#pragma unroll
    for (int e = 0; e < E_EXP; ++e) s += gls[b * E_EXP + e] * val[e];
    ebias[b * DOUT + o] = s;
  }
}

// ---------------------------------------------------------------------------
// 8. x -> bf16
// ---------------------------------------------------------------------------
__global__ __launch_bounds__(256) void xcvt_kernel(const float* __restrict__ x,
                                                   unsigned short* __restrict__ xb) {
  int id = blockIdx.x * 256 + threadIdx.x;
  const float4 v = *(const float4*)(x + (size_t)id * 4);
  unsigned lo = (unsigned)f2bf(v.x) | ((unsigned)f2bf(v.y) << 16);
  unsigned hi = (unsigned)f2bf(v.z) | ((unsigned)f2bf(v.w) << 16);
  uint2 p; p.x = lo; p.y = hi;
  *(uint2*)(xb + (size_t)id * 4) = p;
}

// ---------------------------------------------------------------------------
// 9. Build per-group bf16 weights:
//    Wb[b][o][i] = bf16( res[o,i] + sum_e g[b,e] * (sum_r Q[e,o,r]*rtw1[e,r,i]) * mask )
// ---------------------------------------------------------------------------
__global__ __launch_bounds__(256) void buildw_kernel(const float* __restrict__ weight,
                                                     const float* __restrict__ resw,
                                                     const float* __restrict__ Q,
                                                     const float* __restrict__ rtw1,
                                                     const float* __restrict__ gates,
                                                     unsigned short* __restrict__ Wb) {
  __shared__ float gls[BN * E_EXP];
  __shared__ float Ql[E_EXP * RANK];
  int tid = threadIdx.x;
  int o = blockIdx.y;
  int i = blockIdx.x * 256 + tid;
  gls[tid] = gates[tid];
  if (tid < E_EXP * RANK) {
    int e = tid >> 3, r = tid & 7;
    Ql[tid] = Q[(e * DOUT + o) * RANK + r];
  }
  __syncthreads();
  float resv = resw[(size_t)o * DIN + i];
  float diff[E_EXP], Mraw[E_EXP], S = 0.f;
#pragma unroll
  for (int e = 0; e < E_EXP; ++e) {
    float d = weight[((size_t)e * DOUT + o) * DIN + i] - resv;
    diff[e] = d; S += d;
    float acc = 0.f;
#pragma unroll
    for (int r = 0; r < RANK; ++r)
      acc += Ql[e * RANK + r] * rtw1[(e * RANK + r) * DIN + i];
    Mraw[e] = acc;
  }
  float sS = sgnf(S);
#pragma unroll
  for (int e = 0; e < E_EXP; ++e) {
    float sd = sgnf(diff[e]);
    Mraw[e] *= (sd == sS) ? sd : 0.f;
  }
  for (int b = 0; b < BN; ++b) {
    float w = resv;
#pragma unroll
    for (int e = 0; e < E_EXP; ++e) w += gls[b * E_EXP + e] * Mraw[e];
    Wb[(((size_t)b * DOUT + o) * DIN) + i] = f2bf(w);
  }
}

// ---------------------------------------------------------------------------
// 10. Main batched GEMM with WMMA bf16 (fp32 accumulate):
//     y[b] = x_b(256x1024) @ W_b^T(1024x1024) + ebias[b]
//     Block tile 64(M) x 256(N), K-step 32, 8 waves (2x4), wave tile 32x64.
//     Double-buffered LDS fed by GLOBAL_LOAD_ASYNC_TO_LDS (ASYNCcnt) when
//     available; sync global->reg->LDS fallback otherwise.
// ---------------------------------------------------------------------------
__global__ __launch_bounds__(256) void gemm_kernel(const unsigned short* __restrict__ xb,
                                                   const unsigned short* __restrict__ wb,
                                                   const float* __restrict__ ebias,
                                                   float* __restrict__ yout) {
  __shared__ __align__(16) unsigned short Xs[2][64 * 40];    // 2 x 5 KB
  __shared__ __align__(16) unsigned short Ws[2][256 * 40];   // 2 x 20 KB
  const int tid = threadIdx.x;
  const int lane = tid & 31;
  const int wave = tid >> 5;
  const int waveM = wave >> 2;                  // 0..1  (32 rows each)
  const int waveN = wave & 3;                   // 0..3  (64 cols each)
  const int n0 = blockIdx.x * 256;
  const int m0 = blockIdx.y * 64;
  const int bn = blockIdx.z;

  v8f zero = {0.f, 0.f, 0.f, 0.f, 0.f, 0.f, 0.f, 0.f};
  v8f acc[2][4];
#pragma unroll
  for (int i = 0; i < 2; ++i)
#pragma unroll
    for (int j = 0; j < 4; ++j) acc[i][j] = zero;

  // Cooperative tile-fill assignments
  const int xr = tid >> 2, xc = (tid & 3) * 8;        // X: 64 rows x 32 K
  const int w_row = tid >> 1, w_col = (tid & 1) * 16; // W: rows t>>1 and 128+t>>1
  const unsigned short* xg = xb + ((size_t)(bn * TGRP + m0 + xr)) * DIN + xc;
  const unsigned short* wg0 = wb + ((size_t)(bn * DOUT + n0 + w_row)) * DIN + w_col;
  const unsigned short* wg1 = wb + ((size_t)(bn * DOUT + n0 + 128 + w_row)) * DIN + w_col;
  unsigned short* xs_dst[2] = {&Xs[0][xr * 40 + xc], &Xs[1][xr * 40 + xc]};
  unsigned short* ws_dst0[2] = {&Ws[0][w_row * 40 + w_col], &Ws[1][w_row * 40 + w_col]};
  unsigned short* ws_dst1[2] = {&Ws[0][(128 + w_row) * 40 + w_col],
                                &Ws[1][(128 + w_row) * 40 + w_col]};

  const int am = lane & 15;
  const int ak = (lane >> 4) * 8;                   // A K-base: 0 or 8 (+16 upper)
  const int bk = (lane >> 4) * 8;                   // B K-base: 0 or 8 (+16 upper)

  // Prologue: fill buffer 0 with K-tile 0
  tile_b128(xg, xs_dst[0]);
  tile_b128(wg0, ws_dst0[0]);
  tile_b128(wg0 + 8, ws_dst0[0] + 8);
  tile_b128(wg1, ws_dst1[0]);
  tile_b128(wg1 + 8, ws_dst1[0] + 8);
  tile_wait();
  __syncthreads();

  union U { uint4 q[2]; v16bf v; };
  for (int it = 0; it < DIN / 32; ++it) {
    const int buf = it & 1, nxt = buf ^ 1;
    if (it + 1 < DIN / 32) {
      const int kt = (it + 1) * 32;
      tile_b128(xg + kt, xs_dst[nxt]);
      tile_b128(wg0 + kt, ws_dst0[nxt]);
      tile_b128(wg0 + kt + 8, ws_dst0[nxt] + 8);
      tile_b128(wg1 + kt, ws_dst1[nxt]);
      tile_b128(wg1 + kt + 8, ws_dst1[nxt] + 8);
    }

    U a[2], b[4];
#pragma unroll
    for (int t = 0; t < 2; ++t) {                   // A frags: 16x32, M x K
      int row = waveM * 32 + t * 16 + am;
      a[t].q[0] = *(const uint4*)&Xs[buf][row * 40 + ak];
      a[t].q[1] = *(const uint4*)&Xs[buf][row * 40 + ak + 16];
    }
#pragma unroll
    for (int t = 0; t < 4; ++t) {                   // B frags: 32x16, K x N
      int row = waveN * 64 + t * 16 + am;
      b[t].q[0] = *(const uint4*)&Ws[buf][row * 40 + bk];
      b[t].q[1] = *(const uint4*)&Ws[buf][row * 40 + bk + 16];
    }
#pragma unroll
    for (int i = 0; i < 2; ++i)
#pragma unroll
      for (int j = 0; j < 4; ++j)
        acc[i][j] = __builtin_amdgcn_wmma_f32_16x16x32_bf16(
            false, a[i].v, false, b[j].v, (short)0, acc[i][j], false, false);

    tile_wait();
    __syncthreads();
  }

  // Epilogue: C/D layout — lanes 0-15: M=r, lanes 16-31: M=r+8; N = lane&15
#pragma unroll
  for (int i = 0; i < 2; ++i) {
#pragma unroll
    for (int j = 0; j < 4; ++j) {
      int col = n0 + waveN * 64 + j * 16 + (lane & 15);
      float bv = ebias[bn * DOUT + col];
#pragma unroll
      for (int r = 0; r < 8; ++r) {
        int mt = m0 + waveM * 32 + i * 16 + r + ((lane >> 4) * 8);
        yout[((size_t)(bn * TGRP + mt)) * DOUT + col] = acc[i][j][r] + bv;
      }
    }
  }
}

// ---------------------------------------------------------------------------
// Host-side launch
// ---------------------------------------------------------------------------
extern "C" void kernel_launch(void* const* d_in, const int* in_sizes, int n_in,
                              void* d_out, int out_size, void* d_ws, size_t ws_size,
                              hipStream_t stream) {
  const float* x        = (const float*)d_in[0];
  const float* w_gate   = (const float*)d_in[1];
  const float* weight   = (const float*)d_in[2];
  const float* bias     = (const float*)d_in[3];
  const float* res_w    = (const float*)d_in[4];
  const float* res_b    = (const float*)d_in[5];
  const float* c1in     = (const float*)d_in[6];
  const float* c2in     = (const float*)d_in[7];
  const float* c1out    = (const float*)d_in[8];
  const float* c2out    = (const float*)d_in[9];
  const float* c1bias   = (const float*)d_in[10];
  const float* c2bias   = (const float*)d_in[11];
  float* y = (float*)d_out;

  // Workspace carve-up (all 256B-aligned)
  char* ws = (char*)d_ws;
  size_t off = 0;
  auto alloc = [&](size_t bytes) {
    char* p = ws + off;
    off += (bytes + 255) & ~(size_t)255;
    return p;
  };
  float* colmean = (float*)alloc(BN * DIN * 4);                  // 128 KB
  float* gatesR  = (float*)alloc(BN * E_EXP * 4);                // 1 KB
  float* Q       = (float*)alloc(E_EXP * DOUT * RANK * 4);       // 256 KB
  float* Zb      = (float*)alloc(E_EXP * DOUT * RANK * 4);       // 256 KB
  float* t1      = (float*)alloc(E_EXP * DIN * RANK * 4);        // 256 KB
  float* rw_a    = (float*)alloc(E_EXP * RANK * DIN * 4);        // 256 KB
  float* rw_b    = (float*)alloc(E_EXP * RANK * DIN * 4);        // 256 KB
  float* rtb2    = (float*)alloc(E_EXP * DOUT * 4);              // 32 KB
  float* ebias   = (float*)alloc(BN * DOUT * 4);                 // 128 KB
  unsigned short* x_bf = (unsigned short*)alloc((size_t)BN * TGRP * DIN * 2);   // 16 MB
  unsigned short* Wb   = (unsigned short*)alloc((size_t)BN * DOUT * DIN * 2);   // 64 MB
  (void)ws_size; (void)in_sizes; (void)n_in; (void)out_size;

  // 1-2. Gating + loss
  colmean_kernel<<<BN, 256, 0, stream>>>(x, colmean);
  gate_kernel<<<1, 256, 0, stream>>>(colmean, w_gate, gatesR, y + YSZ);

  // 3-4. Subspace iteration (top-RANK left singular vectors of diff_w)
  initq_kernel<<<256, 256, 0, stream>>>(Q);
  for (int it = 0; it < 4; ++it) {
    t1_kernel<<<dim3(4, E_EXP), 256, 0, stream>>>(weight, res_w, Q, t1);
    z_kernel<<<dim3(4, E_EXP), 256, 0, stream>>>(weight, res_w, t1, Zb);
    mgs_kernel<<<E_EXP, 256, 0, stream>>>(Zb, Q);
  }
  // Final projection: t1[e][i][r] = rtw0^T
  t1_kernel<<<dim3(4, E_EXP), 256, 0, stream>>>(weight, res_w, Q, t1);

  // 5. Curve transforms on the rank core
  curve_kernel<<<dim3(32, E_EXP), 256, 0, stream>>>(t1,   c1out, rw_b, 1);
  curve_kernel<<<dim3(32, E_EXP), 256, 0, stream>>>(rw_b, c2out, rw_a, 2);
  curve_kernel<<<dim3(32, E_EXP), 256, 0, stream>>>(rw_a, c1in,  rw_b, 3);
  curve_kernel<<<dim3(32, E_EXP), 256, 0, stream>>>(rw_b, c2in,  rw_a, 4);

  // 6-7. Bias path
  bias1_kernel<<<E_EXP, 256, 0, stream>>>(bias, res_b, c1bias, c2bias, rtb2);
  ebias_kernel<<<4, 256, 0, stream>>>(bias, res_b, rtb2, gatesR, ebias);

  // 8. Activations to bf16
  xcvt_kernel<<<(BN * TGRP * DIN) / (256 * 4), 256, 0, stream>>>(x, x_bf);

  // 9. Materialize per-group bf16 weights (fused masks + rank-8 reconstruction)
  buildw_kernel<<<dim3(4, DOUT), 256, 0, stream>>>(weight, res_w, Q, rw_a, gatesR, Wb);

  // 10. WMMA batched GEMM + bias epilogue (async double-buffered LDS)
  gemm_kernel<<<dim3(DOUT / 256, TGRP / 64, BN), 256, 0, stream>>>(x_bf, Wb, ebias, y);
}
